// Encoder_29042568856180
// MI455X (gfx1250) — matrix operand order repaired and checked by reading, not compiled, not run on hardware
//
#include <hip/hip_runtime.h>

typedef __attribute__((ext_vector_type(16))) _Float16 v16h;
typedef __attribute__((ext_vector_type(8)))  float    v8f;

#define TT 512
#define BBATCH 256

__device__ __forceinline__ float sigmf(float x) { return 1.0f / (1.0f + __expf(-x)); }

// Packed fragment layout: frag(tile, kk) = 32 lanes x 16 f16 (32 B contiguous
// per lane). Index: (((tile*KT + kk)*32 + lane)*16 + elem).
// A-layout elem e -> kb = (e<8) ? 8*half + e : 16 + 8*half + (e-8)
// B-layout elem e -> kb = (e<8) ? 8*half + e : 16 + 8*half + (e-8)

// ---- Pack X[M,K] (f32) -> A fragments (f16), zero-pad K to 32*KT ----
__global__ void pack_a_kernel(const float* __restrict__ X, _Float16* __restrict__ PA,
                              int K, int KT, int total) {
  int gid = blockIdx.x * 256 + threadIdx.x;   // = (rowtile*KT + kk)*32 + lane
  if (gid >= total) return;
  const int lane    = gid & 31;
  const int tk      = gid >> 5;
  const int kk      = tk % KT;
  const int rowtile = tk / KT;
  const int m       = rowtile * 16 + (lane & 15);
  const int half    = lane >> 4;
  const float* row  = X + (size_t)m * K;
  v16h a;
#pragma unroll
  for (int e = 0; e < 16; ++e) {
    int kb = (e < 8) ? (8 * half + e) : (16 + 8 * half + (e - 8));
    int k  = kk * 32 + kb;
    int kc = (k < K) ? k : (K - 1);          // branch-free: always-in-bounds load
    float v = row[kc];
    a[e] = (_Float16)((k < K) ? v : 0.0f);   // v_cndmask select
  }
  *(v16h*)(PA + (size_t)gid * 16) = a;
}

// ---- Pack W[N,K] (f32) -> B fragments (f16), zero-pad K ----
__global__ void pack_b_kernel(const float* __restrict__ W, _Float16* __restrict__ PB,
                              int K, int KT, int total) {
  int gid = blockIdx.x * 256 + threadIdx.x;   // = (ntile*KT + kk)*32 + lane
  if (gid >= total) return;
  const int lane  = gid & 31;
  const int tk    = gid >> 5;
  const int kk    = tk % KT;
  const int ntile = tk / KT;
  const int n     = ntile * 16 + (lane & 15);
  const int half  = lane >> 4;
  const float* row = W + (size_t)n * K;
  v16h b;
#pragma unroll
  for (int e = 0; e < 16; ++e) {
    int k  = kk * 32 + ((e < 8) ? (8 * half + e) : (16 + 8 * half + (e - 8)));
    int kc = (k < K) ? k : (K - 1);
    float v = row[kc];
    b[e] = (_Float16)((k < K) ? v : 0.0f);
  }
  *(v16h*)(PB + (size_t)gid * 16) = b;
}

// ---- Kernel 1: XG = A @ B^T + bias from packed fragments ----
__global__ void gemm_xg_wmma(const _Float16* __restrict__ PA, const _Float16* __restrict__ PB,
                             const float* __restrict__ bih, const float* __restrict__ bhh,
                             float* __restrict__ XG, int KT, int N) {
  const int lane   = threadIdx.x & 31;
  const int wave   = threadIdx.x >> 5;
  const int ntiles = N >> 4;
  const long tile  = (long)blockIdx.x * 8 + wave;
  const int  mi    = (int)(tile / ntiles);
  const int  ni    = (int)(tile % ntiles);

  v8f c = {};
  for (int kk = 0; kk < KT; ++kk) {
    v16h a = *(const v16h*)(PA + (((size_t)mi * KT + kk) * 32 + lane) * 16);
    v16h b = *(const v16h*)(PB + (((size_t)ni * KT + kk) * 32 + lane) * 16);
    c = __builtin_amdgcn_wmma_f32_16x16x32_f16(false, a, false, b, (short)0, c, false, false);
  }
  const int   n    = (ni << 4) + (lane & 15);
  const float bias = bih[n] + bhh[n];
  const int   mrow = (mi << 4) + (lane >> 4) * 8;
#pragma unroll
  for (int r = 0; r < 8; ++r)   // NT store: xg is a single-use stream
    __builtin_nontemporal_store(c[r] + bias, &XG[(size_t)(mrow + r) * N + n]);
}

// ---- Kernel 2: sequential LSTM scan with WMMA recurrence ----
// Block = (16 hidden units) x (128 batch rows). Whh fragments live in
// REGISTERS for all 512 steps; h history is kept in packed f16 fragment
// layout so A-fragment loads are single contiguous 32B/lane loads.
template <int H>
__global__ void lstm_scan_wmma(const float* __restrict__ XG,    // [T,B,4H]
                               const float* __restrict__ Whh,   // [4H,H]
                               _Float16* __restrict__ HPK,      // packed h frags
                               float* __restrict__ HSF,         // optional f32 h [T,B,H]
                               int* __restrict__ bar) {
  constexpr int KT  = H / 32;
  constexpr int GST = 66;                   // padded LDS row stride (floats)
  __shared__ float lds_g[128 * GST];

  const int lane   = threadIdx.x & 31;
  const int wave   = threadIdx.x >> 5;
  const int utiles = H / 16;
  const int u0     = (blockIdx.x % utiles) * 16;
  const int bbase  = (blockIdx.x / utiles) * 128;
  const int n_l    = lane & 15;
  const int half   = lane >> 4;
  const int mi     = wave;                       // 8 waves -> 8 M-tiles
  const int rbase  = mi * 16 + (lane >> 4) * 8;  // local batch row base of C rows

  // Whh B-fragments -> registers (loop-invariant across all timesteps).
  v16h bfr[4][KT];
#pragma unroll
  for (int q = 0; q < 4; ++q) {
#pragma unroll
    for (int kk = 0; kk < KT; ++kk) {
      const float* row = Whh + (size_t)(q * H + u0 + n_l) * H;
      v16h b;
#pragma unroll
      for (int e = 0; e < 16; ++e)
        b[e] = (_Float16)row[kk * 32 + ((e < 8) ? (8 * half + e) : (16 + 8 * half + (e - 8)))];
      bfr[q][kk] = b;
    }
  }

  float cst[8];
#pragma unroll
  for (int u = 0; u < 8; ++u) cst[u] = 0.0f;
  const int lb = threadIdx.x & 127;   // local batch row for elementwise
  const int uh = threadIdx.x >> 7;    // which 8-unit half
  const int rr = lb & 15;             // row within tile for packed writes

  for (int t = 0; t < TT; ++t) {
    v16h afr[KT];
    if (t > 0) {
      const size_t rowtile = (size_t)(t - 1) * 16 + (bbase >> 4) + mi;
#pragma unroll
      for (int kk = 0; kk < KT; ++kk)
        afr[kk] = *(const v16h*)(HPK + ((rowtile * KT + kk) * 32 + lane) * 16);
    }
#pragma unroll
    for (int q = 0; q < 4; ++q) {
      v8f c = {};
      if (t > 0) {
#pragma unroll
        for (int kk = 0; kk < KT; ++kk)
          c = __builtin_amdgcn_wmma_f32_16x16x32_f16(false, afr[kk], false, bfr[q][kk],
                                                     (short)0, c, false, false);
      }
      // Fused, coalesced +xg add (NT load: stream-once) while spilling C to LDS.
      const float* xcol = XG + ((size_t)t * BBATCH + bbase + rbase) * (4 * H)
                             + (size_t)q * H + u0 + n_l;
#pragma unroll
      for (int r = 0; r < 8; ++r)
        lds_g[(rbase + r) * GST + q * 16 + n_l] =
            c[r] + __builtin_nontemporal_load(xcol + (size_t)r * 4 * H);
    }
    __syncthreads();

    // Elementwise LSTM cell update; c-state in registers; h written straight
    // into packed f16 fragment layout (consumed by WMMA next step / next layer).
    {
      const float* gr = lds_g + lb * GST;
      const size_t rowtile = (size_t)t * 16 + ((bbase + lb) >> 4);
#pragma unroll
      for (int u = 0; u < 8; ++u) {
        int un = uh * 8 + u;
        float iv = sigmf(gr[un]);
        float fv = sigmf(gr[16 + un]);
        float gv = tanhf(gr[32 + un]);
        float ov = sigmf(gr[48 + un]);
        float cc = fv * cst[u] + iv * gv;
        cst[u] = cc;
        float hv = ov * tanhf(cc);

        int k     = u0 + un;            // hidden index = A-matrix K index
        int kk    = k >> 5;
        int kw    = k & 31;
        int hhalf = (kw >> 3) & 1;
        int lnn   = rr + 16 * hhalf;
        int e     = (kw < 16) ? (kw - 8 * hhalf) : (8 + kw - 16 - 8 * hhalf);
        HPK[((rowtile * KT + kk) * 32 + lnn) * 16 + e] = (_Float16)hv;
        if (HSF) HSF[((size_t)t * BBATCH + bbase + lb) * H + k] = hv;
      }
    }

    // Prefetch next step's xg gate segments (each is one 64B line) so the
    // post-barrier critical path hits WGP$ instead of L2/HBM.
    if (t + 1 < TT) {
      const float* xnext = XG + ((size_t)(t + 1) * BBATCH + bbase + lb) * (4 * H) + u0;
      __builtin_prefetch(xnext + (size_t)(uh * 2 + 0) * H, 0, 0);
      __builtin_prefetch(xnext + (size_t)(uh * 2 + 1) * H, 0, 0);
    }

    __threadfence();
    __syncthreads();
    if (t + 1 < TT) {   // device-wide barrier so all blocks see h(t)
      if (threadIdx.x == 0) {
        atomicAdd(bar, 1);
        const int target = (t + 1) * (int)gridDim.x;
        while (__hip_atomic_load(bar, __ATOMIC_ACQUIRE, __HIP_MEMORY_SCOPE_AGENT) < target)
          __builtin_amdgcn_s_sleep(2);
      }
      __syncthreads();
    }
  }
}

// ---- Kernel 3: layer-3 input projection (K=64, N=4) ----
__global__ void xg3_kernel(const float* __restrict__ HS2, const float* __restrict__ Wih3,
                           const float* __restrict__ bih3, const float* __restrict__ bhh3,
                           float* __restrict__ XG3, int TB) {
  int idx = blockIdx.x * 256 + threadIdx.x;
  if (idx >= TB) return;
  const float* xr = HS2 + (size_t)idx * 64;
  float acc0 = bih3[0] + bhh3[0];
  float acc1 = bih3[1] + bhh3[1];
  float acc2 = bih3[2] + bhh3[2];
  float acc3 = bih3[3] + bhh3[3];
#pragma unroll 8
  for (int k = 0; k < 64; ++k) {
    float x = xr[k];
    acc0 += x * Wih3[0 * 64 + k];
    acc1 += x * Wih3[1 * 64 + k];
    acc2 += x * Wih3[2 * 64 + k];
    acc3 += x * Wih3[3 * 64 + k];
  }
  float* o = XG3 + (size_t)idx * 4;
  o[0] = acc0; o[1] = acc1; o[2] = acc2; o[3] = acc3;
}

// ---- Kernel 4: layer-3 scan (H=1): fully parallel across batch ----
__global__ void scan3_kernel(const float* __restrict__ XG3, const float* __restrict__ Whh3,
                             float* __restrict__ OUT) {
  const int b = threadIdx.x;  // 256 threads
  const float w0 = Whh3[0], w1 = Whh3[1], w2 = Whh3[2], w3 = Whh3[3];
  float c = 0.0f, h = 0.0f;
  for (int t = 0; t < TT; ++t) {
    const float* g = XG3 + ((size_t)t * BBATCH + b) * 4;
    float iv = sigmf(g[0] + w0 * h);
    float fv = sigmf(g[1] + w1 * h);
    float gv = tanhf(g[2] + w2 * h);
    float ov = sigmf(g[3] + w3 * h);
    c = fv * c + iv * gv;
    h = ov * tanhf(c);
    OUT[(size_t)t * BBATCH + b] = h;
  }
}

__global__ void zero_bars(int* bars) {
  if (threadIdx.x < 8) bars[threadIdx.x] = 0;
}

extern "C" void kernel_launch(void* const* d_in, const int* in_sizes, int n_in,
                              void* d_out, int out_size, void* d_ws, size_t ws_size,
                              hipStream_t stream) {
  const float* x    = (const float*)d_in[0];
  const float* Wih1 = (const float*)d_in[1];
  const float* Whh1 = (const float*)d_in[2];
  const float* bih1 = (const float*)d_in[3];
  const float* bhh1 = (const float*)d_in[4];
  const float* Wih2 = (const float*)d_in[5];
  const float* Whh2 = (const float*)d_in[6];
  const float* bih2 = (const float*)d_in[7];
  const float* bhh2 = (const float*)d_in[8];
  const float* Wih3 = (const float*)d_in[9];
  const float* Whh3 = (const float*)d_in[10];
  const float* bih3 = (const float*)d_in[11];
  const float* bhh3 = (const float*)d_in[12];

  const int TB       = TT * BBATCH;   // 131072 rows
  const int ROWTILES = TB / 16;       // 8192

  // ---- workspace layout ----
  float* ws = (float*)d_ws;
  size_t off = 0;
  float* xgA  = ws + off; off += (size_t)TB * 512;   // xg1 -> xg2 -> xg3 (reuse)
  float* hs2f = ws + off; off += (size_t)TB * 64;    // f32 h2 for layer-3 proj
  int*   bars = (int*)(ws + off); off += 16;
  _Float16* h16 = (_Float16*)(ws + off);
  size_t o16 = 0;
  _Float16* PA1 = h16 + o16; o16 += (size_t)ROWTILES * 4 * 512;  // x packed (K 97->128)
  _Float16* PB1 = h16 + o16; o16 += (size_t)32 * 4 * 512;        // Wih1 packed
  _Float16* PB2 = h16 + o16; o16 += (size_t)16 * 4 * 512;        // Wih2 packed
  _Float16* hp1 = h16 + o16; o16 += (size_t)ROWTILES * 4 * 512;  // h1 packed (K=128)
  _Float16* hp2 = h16 + o16; o16 += (size_t)ROWTILES * 2 * 512;  // h2 packed (K=64)

  float* xg1 = xgA;
  float* xg2 = xgA;   // xg1 dead after scan1
  float* xg3 = xgA;   // xg2 dead after scan2

  zero_bars<<<dim3(1), dim3(32), 0, stream>>>(bars);

  // Pack inputs/weights into WMMA fragment layout (f16).
  {
    int totA = ROWTILES * 4 * 32;   // 1,048,576 threads
    pack_a_kernel<<<dim3(totA / 256), dim3(256), 0, stream>>>(x, PA1, 97, 4, totA);
    int totB1 = 32 * 4 * 32;
    pack_b_kernel<<<dim3((totB1 + 255) / 256), dim3(256), 0, stream>>>(Wih1, PB1, 97, 4, totB1);
    int totB2 = 16 * 4 * 32;
    pack_b_kernel<<<dim3((totB2 + 255) / 256), dim3(256), 0, stream>>>(Wih2, PB2, 128, 4, totB2);
  }

  // Layer 1: input GEMM [131072 x 97pad128] x [512 x 97pad128]^T
  gemm_xg_wmma<<<dim3(ROWTILES * 32 / 8), dim3(256), 0, stream>>>(
      PA1, PB1, bih1, bhh1, xg1, 4, 512);
  // Layer 1 scan: 8 unit-chunks x 2 batch-chunks = 16 blocks
  lstm_scan_wmma<128><<<dim3((128 / 16) * 2), dim3(256), 0, stream>>>(
      xg1, Whh1, hp1, (float*)nullptr, bars + 0);

  // Layer 2: input GEMM [131072 x 128] x [256 x 128]^T (A = packed h1)
  gemm_xg_wmma<<<dim3(ROWTILES * 16 / 8), dim3(256), 0, stream>>>(
      hp1, PB2, bih2, bhh2, xg2, 4, 256);
  // Layer 2 scan: 4 unit-chunks x 2 batch-chunks = 8 blocks (also writes f32 h2)
  lstm_scan_wmma<64><<<dim3((64 / 16) * 2), dim3(256), 0, stream>>>(
      xg2, Whh2, hp2, hs2f, bars + 1);

  // Layer 3: tiny projection + fully parallel scan
  xg3_kernel<<<dim3(TB / 256), dim3(256), 0, stream>>>(hs2f, Wih3, bih3, bhh3, xg3, TB);
  scan3_kernel<<<dim3(1), dim3(256), 0, stream>>>(xg3, Whh3, (float*)d_out);
}
